// PFIdentityGNN_13374528160092
// MI455X (gfx1250) — matrix operand order repaired and checked by reading, not compiled, not run on hardware
//
#include <hip/hip_runtime.h>

// ---------------------------------------------------------------------------
// GNN message passing on MI455X (gfx1250, wave32, WMMA f16->f32).
//
// d_in layout assumption (insertion-order flattening of setup_inputs()):
//   0: x [N,4] f32   1: edge_index [2,E] i32   2: edge_attr [E,2] f32
//   3: edge_id (unused)   4: node_emb [N,8] f32   5: edge_emb [E,4] f32
//   6..11 : phi0    W0,b0,W1,b1,W2,b2
//   12..35: psi[l]  W0,b0,W1,b1,W2,b2   l=0..3
//   36..59: upd[l]  W0,b0,W1,b1,W2,b2   l=0..3
//   60..65: readout W0,b0,W1,b1,W2,b2
// ---------------------------------------------------------------------------

#define NN      50000
#define NE      1600000
#define HDIM    64
#define NTILE_N 3125      // NN/16
#define NBLK_N  391       // ceil(3125/8)

typedef __attribute__((ext_vector_type(16))) _Float16 v16h;
typedef __attribute__((ext_vector_type(8)))  _Float16 v8h;
typedef __attribute__((ext_vector_type(8)))  float    v8f;

// ---- f16 transposed weight pool offsets (units: _Float16) -----------------
#define PHI0_W0   0          // kpad 32
#define PHI0_W1   2048       // kpad 64
#define PHI0_W2   6144
#define PSI_BASE  10240      // per layer: W0(kpad96)@+0 W1@+6144 W2@+10240
#define PSI_STRIDE 14336
#define UPD_BASE  67584      // per layer: W0(kpad160)@+0 W1@+10240 W2@+14336
#define UPD_STRIDE 18432
#define RO_W0     141312
#define RO_W1     145408

// ---- workspace byte offsets ----------------------------------------------
#define WS_POOL 0
#define WS_HA   (512*1024)
#define WS_HB   (WS_HA + (size_t)NN*HDIM*2)
#define WS_M    (WS_HB + (size_t)NN*HDIM*2)

__device__ __forceinline__ v8f wmma16(v16h a, v16h b, v8f c) {
  return __builtin_amdgcn_wmma_f32_16x16x32_f16(false, a, false, b,
                                                (short)0, c, false, false);
}

// ReLU as a single v_med3_f32 (x finite: med3(x, 0, +MAX) == max(x, 0)).
__device__ __forceinline__ float relu1(float f) {
  return __builtin_amdgcn_fmed3f(f, 0.0f, 3.402823466e+38f);
}

// A-fragment (16x32 f16) from LDS, row-major tile with stride sx halves.
// ISA layout: lane L holds row L&15; K = (L>>4)*8 + {0..7} and +16..+23.
__device__ __forceinline__ v16h ldA(const _Float16* X, int sx, int kc, int lane) {
  const int r = lane & 15, p = lane >> 4;
  const _Float16* q = X + r * sx + kc * 32 + p * 8;
  v8h lo = *(const v8h*)(q);
  v8h hi = *(const v8h*)(q + 16);
  v16h a;
#pragma unroll
  for (int i = 0; i < 8; ++i) { a[i] = lo[i]; a[i + 8] = hi[i]; }
  return a;
}

// B-fragment (32x16 f16) from prepacked Wt[dout][ldk] (transposed weights).
// ISA layout: lane L holds column L&15; K = (L>>4)*8 + {0..7} and +16..+23.
__device__ __forceinline__ v16h ldB(const _Float16* Wt, int ldk, int n0,
                                    int kc, int lane) {
  const int n = n0 + (lane & 15);
  const _Float16* q = Wt + (size_t)n * ldk + kc * 32 + (lane >> 4) * 8;
  v8h lo = *(const v8h*)(q);
  v8h hi = *(const v8h*)(q + 16);
  v16h b;
#pragma unroll
  for (int i = 0; i < 8; ++i) { b[i] = lo[i]; b[i + 8] = hi[i]; }
  return b;
}

// Generic M=16*RT x 64 layer with A fragments cached in registers:
// A is read from LDS once (RT*NKC frags), each B fragment feeds RT WMMAs.
// emit(nt, rt, acc) consumes one 16x16 f32 result tile.
template <int NKC, int RT, typename F>
__device__ __forceinline__ void layer_rt(const _Float16* X, int sx,
                                         const _Float16* Wt,
                                         const float* __restrict__ bias,
                                         int lane, F&& emit) {
  v16h a[RT][NKC];
#pragma unroll
  for (int rt = 0; rt < RT; ++rt)
#pragma unroll
    for (int kc = 0; kc < NKC; ++kc)
      a[rt][kc] = ldA(X + rt * 16 * sx, sx, kc, lane);
#pragma unroll
  for (int nt = 0; nt < 4; ++nt) {
    const float bv = bias[nt * 16 + (lane & 15)];
    v8f acc[RT];
#pragma unroll
    for (int rt = 0; rt < RT; ++rt)
#pragma unroll
      for (int i = 0; i < 8; ++i) acc[rt][i] = bv;
#pragma unroll
    for (int kc = 0; kc < NKC; ++kc) {
      v16h b = ldB(Wt, NKC * 32, nt * 16, kc, lane);
#pragma unroll
      for (int rt = 0; rt < RT; ++rt) acc[rt] = wmma16(a[rt][kc], b, acc[rt]);
    }
#pragma unroll
    for (int rt = 0; rt < RT; ++rt) emit(nt, rt, acc[rt]);
  }
}

// 16-row layer -> LDS f16 (optional relu).
template <int NKC>
__device__ __forceinline__ void mlp_layer16(const _Float16* X, int sx,
                                            const _Float16* Wt,
                                            const float* __restrict__ bias,
                                            _Float16* H, int sh, int lane,
                                            bool relu) {
  const int rb = (lane >> 4) * 8, cl = lane & 15;
  layer_rt<NKC, 1>(X, sx, Wt, bias, lane,
                   [&](int nt, int, v8f acc) {
                     const int col = nt * 16 + cl;
#pragma unroll
                     for (int v = 0; v < 8; ++v) {
                       float f = acc[v];
                       if (relu) f = relu1(f);
                       H[(rb + v) * sh + col] = (_Float16)f;
                     }
                   });
}

// Final 16-row layer (K=64) -> global h (f16 row-major), no relu.
__device__ __forceinline__ void mlp_layer16_out(const _Float16* X, int sx,
                                                const _Float16* Wt,
                                                const float* __restrict__ bias,
                                                _Float16* __restrict__ hout,
                                                int nbase, int lane) {
  const int rb = (lane >> 4) * 8, cl = lane & 15;
  layer_rt<2, 1>(X, sx, Wt, bias, lane,
                 [&](int nt, int, v8f acc) {
                   const int col = nt * 16 + cl;
#pragma unroll
                   for (int v = 0; v < 8; ++v)
                     hout[(size_t)(nbase + rb + v) * HDIM + col] =
                         (_Float16)acc[v];
                 });
}

// ---------------------------------------------------------------------------
// Weight prep: f32 [din,64] row-major -> f16 transposed [64, kpad], zero-pad K.
struct PrepTable {
  const float* src[29];
  int din[29];
  int kpad[29];
  int dstoff[29];
};

__global__ void prep_weights_kernel(PrepTable t, _Float16* __restrict__ pool) {
  const int e = blockIdx.x;
  const float* __restrict__ src = t.src[e];
  const int din = t.din[e], kpad = t.kpad[e], off = t.dstoff[e];
  const int tot = 64 * kpad;
  for (int i = threadIdx.x; i < tot; i += blockDim.x) {
    const int n = i / kpad, k = i - n * kpad;
    pool[off + i] = (k < din) ? (_Float16)src[(size_t)k * 64 + n]
                              : (_Float16)0.f;
  }
}

__global__ void zero_m_kernel(float4* __restrict__ m) {
  const size_t i = (size_t)blockIdx.x * blockDim.x + threadIdx.x;
  m[i] = make_float4(0.f, 0.f, 0.f, 0.f);   // grid covers NN*64/4 exactly
}

// ---------------------------------------------------------------------------
// phi0: h = MLP([x, z]); 12 -> pad 32 -> 64 -> 64 -> 64. One 16-node tile/wave.
__global__ void phi0_kernel(const float* __restrict__ x,
                            const float* __restrict__ z,
                            const _Float16* __restrict__ w0,
                            const _Float16* __restrict__ w1,
                            const _Float16* __restrict__ w2,
                            const float* __restrict__ b0,
                            const float* __restrict__ b1,
                            const float* __restrict__ b2,
                            _Float16* __restrict__ hout) {
  __shared__ _Float16 sX[8][16 * 40];
  __shared__ _Float16 sH1[8][16 * 72];
  __shared__ _Float16 sH2[8][16 * 72];
  const int lane = threadIdx.x & 31, wave = threadIdx.x >> 5;
  const int tile = blockIdx.x * 8 + wave;
  if (tile >= NTILE_N) return;
  _Float16* X = sX[wave];
  const int r = lane & 15, part = lane >> 4;
  const int n = tile * 16 + r;
  if (part == 0) {
    _Float16* xr = X + r * 40;
#pragma unroll
    for (int j = 0; j < 4; ++j) xr[j] = (_Float16)x[(size_t)n * 4 + j];
#pragma unroll
    for (int j = 0; j < 8; ++j) xr[4 + j] = (_Float16)z[(size_t)n * 8 + j];
  } else {
    for (int j = 12; j < 32; ++j) X[r * 40 + j] = (_Float16)0.f;
  }
  asm volatile("s_wait_dscnt 0" ::: "memory");
  mlp_layer16<1>(X, 40, w0, b0, sH1[wave], 72, lane, true);
  asm volatile("s_wait_dscnt 0" ::: "memory");
  mlp_layer16<2>(sH1[wave], 72, w1, b1, sH2[wave], 72, lane, true);
  asm volatile("s_wait_dscnt 0" ::: "memory");
  mlp_layer16_out(sH2[wave], 72, w2, b2, hout, tile * 16, lane);
}

// ---------------------------------------------------------------------------
// Edge messages: msg = MLP([h[src], edge_attr, edge_emb]); scatter-add to m.
// One 64-edge tile per wave (four 16-row WMMA tiles share every B fragment),
// A fragments register-cached so LDS is read once per layer.
__global__ void edge_msg_kernel(const int* __restrict__ ei,
                                const float* __restrict__ eattr,
                                const float* __restrict__ eemb,
                                const _Float16* __restrict__ h,
                                const _Float16* __restrict__ w0,
                                const _Float16* __restrict__ w1,
                                const _Float16* __restrict__ w2,
                                const float* __restrict__ b0,
                                const float* __restrict__ b1,
                                const float* __restrict__ b2,
                                float* __restrict__ m) {
  __shared__ _Float16 sX[2][64 * 104];   // 70 -> pad 96, stride 104
  __shared__ _Float16 sH[2][64 * 72];
  __shared__ int sDst[2][64];
  const int lane = threadIdx.x & 31, wave = threadIdx.x >> 5;
  const size_t e0 = ((size_t)blockIdx.x * 2 + wave) * 64; // 12500*2*64 = NE
  _Float16* X = sX[wave];
  _Float16* H = sH[wave];
  int* dstb = sDst[wave];

  // ---- gather: two edges per lane -------------------------------------
#pragma unroll
  for (int half = 0; half < 2; ++half) {
    const int ee = half * 32 + lane;
    const size_t eg = e0 + ee;
    const int src = ei[eg];
    dstb[ee] = ei[NE + eg];
    const uint4* hp = (const uint4*)(h + (size_t)src * HDIM);
    uint4* xp = (uint4*)(X + ee * 104);
#pragma unroll
    for (int j = 0; j < 8; ++j) xp[j] = hp[j];          // 64 halves of h
    const float* ea = eattr + eg * 2;
    const float* ee4 = eemb + eg * 4;
    _Float16* xr = X + ee * 104;
    xr[64] = (_Float16)ea[0];
    xr[65] = (_Float16)ea[1];
#pragma unroll
    for (int j = 0; j < 4; ++j) xr[66 + j] = (_Float16)ee4[j];
    for (int j = 70; j < 96; ++j) xr[j] = (_Float16)0.f; // K padding
  }
  asm volatile("s_wait_dscnt 0" ::: "memory");

  const int rb = (lane >> 4) * 8, cl = lane & 15;

  // ---- layer 1: X(64x96) x W0t, relu -> H ------------------------------
  layer_rt<3, 4>(X, 104, w0, b0, lane,
                 [&](int nt, int rt, v8f acc) {
                   const int col = nt * 16 + cl, rr = rt * 16 + rb;
#pragma unroll
                   for (int v = 0; v < 8; ++v)
                     H[(rr + v) * 72 + col] = (_Float16)relu1(acc[v]);
                 });
  asm volatile("s_wait_dscnt 0" ::: "memory");

  // ---- layer 2: H(64x64) x W1t, relu -> X (reused, stride 104) ---------
  layer_rt<2, 4>(H, 72, w1, b1, lane,
                 [&](int nt, int rt, v8f acc) {
                   const int col = nt * 16 + cl, rr = rt * 16 + rb;
#pragma unroll
                   for (int v = 0; v < 8; ++v)
                     X[(rr + v) * 104 + col] = (_Float16)relu1(acc[v]);
                 });
  asm volatile("s_wait_dscnt 0" ::: "memory");

  // ---- layer 3: X(64x64) x W2t -> atomic scatter into m[dst] -----------
  layer_rt<2, 4>(X, 104, w2, b2, lane,
                 [&](int nt, int rt, v8f acc) {
                   const int col = nt * 16 + cl, rr = rt * 16 + rb;
#pragma unroll
                   for (int v = 0; v < 8; ++v)
                     atomicAdd(&m[(size_t)dstb[rr + v] * HDIM + col], acc[v]);
                 });
}

// ---------------------------------------------------------------------------
// Node update: h' = MLP([h, m, z]); 136 -> pad 160 -> 64 -> 64 -> 64.
__global__ void upd_kernel(const _Float16* __restrict__ hcur,
                           const float* __restrict__ m,
                           const float* __restrict__ z,
                           const _Float16* __restrict__ w0,
                           const _Float16* __restrict__ w1,
                           const _Float16* __restrict__ w2,
                           const float* __restrict__ b0,
                           const float* __restrict__ b1,
                           const float* __restrict__ b2,
                           _Float16* __restrict__ hout) {
  __shared__ _Float16 sX[8][16 * 168];   // 136 -> pad 160, stride 168
  __shared__ _Float16 sH[8][16 * 72];
  const int lane = threadIdx.x & 31, wave = threadIdx.x >> 5;
  const int tile = blockIdx.x * 8 + wave;
  if (tile >= NTILE_N) return;
  _Float16* X = sX[wave];
  const int r = lane & 15, part = lane >> 4;
  const int n = tile * 16 + r;
  {
    const uint4* hp = (const uint4*)(hcur + (size_t)n * HDIM + part * 32);
    uint4* xp = (uint4*)(X + r * 168 + part * 32);
#pragma unroll
    for (int j = 0; j < 4; ++j) xp[j] = hp[j];          // 32 halves of h
    const float4* mp = (const float4*)(m + (size_t)n * HDIM + part * 32);
    _Float16* q = X + r * 168 + 64 + part * 32;
#pragma unroll
    for (int j = 0; j < 8; ++j) {
      float4 t = mp[j];
      q[j * 4 + 0] = (_Float16)t.x; q[j * 4 + 1] = (_Float16)t.y;
      q[j * 4 + 2] = (_Float16)t.z; q[j * 4 + 3] = (_Float16)t.w;
    }
    if (part == 0) {
#pragma unroll
      for (int j = 0; j < 8; ++j)
        X[r * 168 + 128 + j] = (_Float16)z[(size_t)n * 8 + j];
    } else {
      for (int j = 136; j < 160; ++j) X[r * 168 + j] = (_Float16)0.f;
    }
  }
  asm volatile("s_wait_dscnt 0" ::: "memory");
  mlp_layer16<5>(X, 168, w0, b0, sH[wave], 72, lane, true);
  asm volatile("s_wait_dscnt 0" ::: "memory");
  mlp_layer16<2>(sH[wave], 72, w1, b1, X, 168, lane, true);  // reuse X
  asm volatile("s_wait_dscnt 0" ::: "memory");
  mlp_layer16_out(X, 168, w2, b2, hout, tile * 16, lane);
}

// ---------------------------------------------------------------------------
// Readout: out = MLP(h) with final 64 -> 1 done as a per-node dot product.
__global__ void readout_kernel(const _Float16* __restrict__ h,
                               const _Float16* __restrict__ w0,
                               const _Float16* __restrict__ w1,
                               const float* __restrict__ w2,   // [64] f32
                               const float* __restrict__ b0,
                               const float* __restrict__ b1,
                               const float* __restrict__ b2,   // [1]
                               float* __restrict__ out) {
  __shared__ _Float16 sX[8][16 * 72];
  __shared__ _Float16 sH[8][16 * 72];
  const int lane = threadIdx.x & 31, wave = threadIdx.x >> 5;
  const int tile = blockIdx.x * 8 + wave;
  if (tile >= NTILE_N) return;
  _Float16* X = sX[wave];
  const int r = lane & 15, part = lane >> 4;
  const int n = tile * 16 + r;
  {
    const uint4* hp = (const uint4*)(h + (size_t)n * HDIM + part * 32);
    uint4* xp = (uint4*)(X + r * 72 + part * 32);
#pragma unroll
    for (int j = 0; j < 4; ++j) xp[j] = hp[j];
  }
  asm volatile("s_wait_dscnt 0" ::: "memory");
  mlp_layer16<2>(X, 72, w0, b0, sH[wave], 72, lane, true);
  asm volatile("s_wait_dscnt 0" ::: "memory");
  mlp_layer16<2>(sH[wave], 72, w1, b1, X, 72, lane, true);   // reuse X
  asm volatile("s_wait_dscnt 0" ::: "memory");
  if (lane < 16) {
    float acc = b2[0];
#pragma unroll 8
    for (int c = 0; c < 64; ++c) acc += (float)X[lane * 72 + c] * w2[c];
    out[tile * 16 + lane] = acc;
  }
}

// ---------------------------------------------------------------------------
extern "C" void kernel_launch(void* const* d_in, const int* in_sizes, int n_in,
                              void* d_out, int out_size, void* d_ws,
                              size_t ws_size, hipStream_t stream) {
  (void)in_sizes; (void)n_in; (void)out_size; (void)ws_size;
  const float* x     = (const float*)d_in[0];
  const int*   ei    = (const int*)d_in[1];
  const float* eattr = (const float*)d_in[2];
  const float* nemb  = (const float*)d_in[4];
  const float* eemb  = (const float*)d_in[5];

  _Float16* pool = (_Float16*)((char*)d_ws + WS_POOL);
  _Float16* hA   = (_Float16*)((char*)d_ws + WS_HA);
  _Float16* hB   = (_Float16*)((char*)d_ws + WS_HB);
  float*    m    = (float*)((char*)d_ws + WS_M);

  // ---- weight prep table ----
  PrepTable t;
  int e = 0;
  auto put = [&](int idx, int din, int kpad, int off) {
    t.src[e] = (const float*)d_in[idx];
    t.din[e] = din; t.kpad[e] = kpad; t.dstoff[e] = off; ++e;
  };
  put(6, 12, 32, PHI0_W0);
  put(8, 64, 64, PHI0_W1);
  put(10, 64, 64, PHI0_W2);
  for (int l = 0; l < 4; ++l) {
    put(12 + 6 * l, 70, 96, PSI_BASE + l * PSI_STRIDE);
    put(14 + 6 * l, 64, 64, PSI_BASE + l * PSI_STRIDE + 6144);
    put(16 + 6 * l, 64, 64, PSI_BASE + l * PSI_STRIDE + 10240);
  }
  for (int l = 0; l < 4; ++l) {
    put(36 + 6 * l, 136, 160, UPD_BASE + l * UPD_STRIDE);
    put(38 + 6 * l, 64, 64, UPD_BASE + l * UPD_STRIDE + 10240);
    put(40 + 6 * l, 64, 64, UPD_BASE + l * UPD_STRIDE + 14336);
  }
  put(60, 64, 64, RO_W0);
  put(62, 64, 64, RO_W1);

  prep_weights_kernel<<<29, 256, 0, stream>>>(t, pool);

  phi0_kernel<<<NBLK_N, 256, 0, stream>>>(
      x, nemb, pool + PHI0_W0, pool + PHI0_W1, pool + PHI0_W2,
      (const float*)d_in[7], (const float*)d_in[9], (const float*)d_in[11], hA);

  _Float16* hc = hA;
  _Float16* hn = hB;
  for (int l = 0; l < 4; ++l) {
    zero_m_kernel<<<3125, 256, 0, stream>>>((float4*)m);
    const _Float16* pw = pool + PSI_BASE + l * PSI_STRIDE;
    edge_msg_kernel<<<12500, 64, 0, stream>>>(
        ei, eattr, eemb, hc, pw, pw + 6144, pw + 10240,
        (const float*)d_in[13 + 6 * l], (const float*)d_in[15 + 6 * l],
        (const float*)d_in[17 + 6 * l], m);
    const _Float16* uw = pool + UPD_BASE + l * UPD_STRIDE;
    upd_kernel<<<NBLK_N, 256, 0, stream>>>(
        hc, m, nemb, uw, uw + 10240, uw + 14336,
        (const float*)d_in[37 + 6 * l], (const float*)d_in[39 + 6 * l],
        (const float*)d_in[41 + 6 * l], hn);
    _Float16* tmp = hc; hc = hn; hn = tmp;
  }

  readout_kernel<<<NBLK_N, 256, 0, stream>>>(
      hc, pool + RO_W0, pool + RO_W1, (const float*)d_in[64],
      (const float*)d_in[61], (const float*)d_in[63], (const float*)d_in[65],
      (float*)d_out);
}
